// ViTCA_4956392259622
// MI455X (gfx1250) — matrix operand order repaired and checked by reading, not compiled, not run on hardware
//
#include <hip/hip_runtime.h>
#include <hip/hip_bf16.h>
#include <math.h>

// ---------------------------------------------------------------------------
// ViT-CA forward for gfx1250 (MI455X). Big GEMMs on v_wmma_f32_16x16x32_f16,
// localized 3x3 attention as wave32 shuffle-reduce, fused LN / GELU / residual
// epilogues. f32 accumulation everywhere; activations cast to f16 post-LN.
// ---------------------------------------------------------------------------

typedef _Float16 half_t;
typedef __attribute__((ext_vector_type(8)))  _Float16 v8h;
typedef __attribute__((ext_vector_type(16))) _Float16 v16h;
typedef __attribute__((ext_vector_type(8)))  float    v8f;

#define HEADS 8
#define HEAD_DIM 32
#define DIM 256
#define MLP_DIM 1024
#define NTOK 32768          // BATCH * 64 * 64
#define NPOS 4096
#define BATCH 8
#define CELL_DIM 15
#define UPDATE_DIM 12
#define PE_IN 3

// ---------------------------------------------------------------------------
// helpers
// ---------------------------------------------------------------------------
__device__ __forceinline__ float wave_sum(float v) {
    #pragma unroll
    for (int off = 16; off; off >>= 1) v += __shfl_xor(v, off, 32);
    return v;
}

__device__ __forceinline__ v16h load_a16(const half_t* p) {
    // p points at 8 contiguous halfs; p+16 at 8 more (K and K+16 chunks)
    v8h lo = *(const v8h*)p;
    v8h hi = *(const v8h*)(p + 16);
    return __builtin_shufflevector(lo, hi, 0,1,2,3,4,5,6,7,8,9,10,11,12,13,14,15);
}

__device__ __forceinline__ v16h load_b16(const half_t* p) {
    // 16 contiguous halfs
    v8h lo = *(const v8h*)p;
    v8h hi = *(const v8h*)(p + 8);
    return __builtin_shufflevector(lo, hi, 0,1,2,3,4,5,6,7,8,9,10,11,12,13,14,15);
}

__device__ __forceinline__ v8f wmma16(v16h a, v16h b, v8f c) {
    return __builtin_amdgcn_wmma_f32_16x16x32_f16(false, a, false, b, (short)0, c,
                                                  false, false);
}

// ---------------------------------------------------------------------------
// weight cast + transpose: dst[o*K + k] = (f16) src[k*N + o]
// ---------------------------------------------------------------------------
__global__ void castT_kernel(const float* __restrict__ src, half_t* __restrict__ dst,
                             int K, int N) {
    int t = blockIdx.x * blockDim.x + threadIdx.x;
    if (t >= K * N) return;
    int k = t / N;
    int o = t - k * N;
    dst[(size_t)o * K + k] = (half_t)src[(size_t)k * N + o];
}

// ---------------------------------------------------------------------------
// embed: x[b,pos,:] = cells[b,:,pos] @ We + be + PE(pos)
// one thread per (token, 4 dims)
// ---------------------------------------------------------------------------
__global__ void embed_kernel(const float* __restrict__ cells,
                             const float* __restrict__ We,
                             const float* __restrict__ be,
                             float* __restrict__ x) {
    int tid = blockIdx.x * blockDim.x + threadIdx.x;    // NTOK*64 threads
    int token = tid >> 6;
    int d4 = (tid & 63) << 2;
    int pos = token & (NPOS - 1);
    int b = token >> 12;

    float a0 = be[d4], a1 = be[d4 + 1], a2 = be[d4 + 2], a3 = be[d4 + 3];
    #pragma unroll
    for (int c = 0; c < CELL_DIM; ++c) {
        float cv = cells[((size_t)(b * CELL_DIM + c)) * NPOS + pos];
        const float4 w = *(const float4*)(We + c * DIM + d4);
        a0 += cv * w.x; a1 += cv * w.y; a2 += cv * w.z; a3 += cv * w.w;
    }
    // positional encoding: pe[pos][2i]=sin(pos*div_i), pe[pos][2i+1]=cos(...)
    const float kf = -9.210340371976184f / 256.0f;   // -ln(10000)/DIM
    float fp = (float)pos;
    int i0 = d4 >> 1;                                 // pair index
    float div0 = expf((float)(2 * i0) * kf);
    float div1 = expf((float)(2 * (i0 + 1)) * kf);
    a0 += sinf(fp * div0); a1 += cosf(fp * div0);
    a2 += sinf(fp * div1); a3 += cosf(fp * div1);

    float4 r; r.x = a0; r.y = a1; r.z = a2; r.w = a3;
    *(float4*)(x + (size_t)token * DIM + d4) = r;
}

// ---------------------------------------------------------------------------
// layernorm + f16 cast: one wave per token, lane owns 8 contiguous dims
// ---------------------------------------------------------------------------
__global__ void ln_cast_kernel(const float* __restrict__ x,
                               const float* __restrict__ g,
                               const float* __restrict__ bt,
                               half_t* __restrict__ y) {
    int lane = threadIdx.x & 31;
    int tok = (blockIdx.x * blockDim.x + threadIdx.x) >> 5;
    int d0 = lane * 8;
    const float4* xp = (const float4*)(x + (size_t)tok * DIM + d0);
    float4 u0 = xp[0], u1 = xp[1];
    float v[8] = {u0.x, u0.y, u0.z, u0.w, u1.x, u1.y, u1.z, u1.w};
    float s = 0.f, q = 0.f;
    #pragma unroll
    for (int i = 0; i < 8; ++i) { s += v[i]; q += v[i] * v[i]; }
    s = wave_sum(s); q = wave_sum(q);
    float mean = s * (1.0f / DIM);
    float var = q * (1.0f / DIM) - mean * mean;
    float rstd = rsqrtf(var + 1e-5f);
    #pragma unroll
    for (int i = 0; i < 8; ++i) {
        int d = d0 + i;
        y[(size_t)tok * DIM + d] = (half_t)((v[i] - mean) * rstd * g[d] + bt[d]);
    }
}

// ---------------------------------------------------------------------------
// WMMA GEMM: C[M,N] = A[M,K](f16 row-major) x Bt[N,K](f16, B transposed)
// block = 256 threads = 8 waves (2x4), wave tile 32x32, block tile 64x128.
// EPI 0: C=acc (f32)
// EPI 1: C=acc + bias[col] + C  (in-place residual)
// EPI 2: H=(f16) gelu(acc + bias[col])
// ---------------------------------------------------------------------------
template <int EPI>
__global__ __launch_bounds__(256) void gemm_wmma_kernel(
    const half_t* __restrict__ A, const half_t* __restrict__ Bt,
    const float* __restrict__ bias, float* __restrict__ C,
    half_t* __restrict__ H, int M, int N, int K) {
    const int lane = threadIdx.x & 31;
    const int warp = threadIdx.x >> 5;
    const int r16 = lane & 15;
    const int hs = lane >> 4;
    const int rowBase = blockIdx.y * 64 + (warp & 1) * 32;
    const int colBase = blockIdx.x * 128 + (warp >> 1) * 32;

    v8f acc00 = {}, acc01 = {}, acc10 = {}, acc11 = {};

    const half_t* ar0 = A + (size_t)(rowBase + r16) * K;
    const half_t* ar1 = A + (size_t)(rowBase + 16 + r16) * K;
    const half_t* br0 = Bt + (size_t)(colBase + r16) * K;
    const half_t* br1 = Bt + (size_t)(colBase + 16 + r16) * K;

    for (int k0 = 0; k0 < K; k0 += 32) {
        v16h a0 = load_a16(ar0 + k0 + hs * 8);
        v16h a1 = load_a16(ar1 + k0 + hs * 8);
        v16h b0 = load_a16(br0 + k0 + hs * 8);
        v16h b1 = load_a16(br1 + k0 + hs * 8);
        acc00 = wmma16(a0, b0, acc00);
        acc01 = wmma16(a0, b1, acc01);
        acc10 = wmma16(a1, b0, acc10);
        acc11 = wmma16(a1, b1, acc11);
    }

    #pragma unroll
    for (int mi = 0; mi < 2; ++mi) {
        #pragma unroll
        for (int ni = 0; ni < 2; ++ni) {
            v8f acc = (mi == 0) ? (ni == 0 ? acc00 : acc01)
                                : (ni == 0 ? acc10 : acc11);
            int col = colBase + ni * 16 + r16;
            #pragma unroll
            for (int j = 0; j < 8; ++j) {
                int row = rowBase + mi * 16 + hs * 8 + j;
                size_t idx = (size_t)row * N + col;
                float v = acc[j];
                if constexpr (EPI == 0) {
                    C[idx] = v;
                } else if constexpr (EPI == 1) {
                    C[idx] = v + bias[col] + C[idx];
                } else {
                    float t = v + bias[col];
                    H[idx] = (half_t)(0.5f * t * (1.0f + erff(t * 0.70710678118654752f)));
                }
            }
        }
    }
}

// ---------------------------------------------------------------------------
// localized attention: one wave per (batch, pos, head). lane = head dim.
// qkv[token, 0:256]=q, [256:512]=k, [512:768]=v. 3x3 circular neighborhood.
// ---------------------------------------------------------------------------
__global__ void attn_kernel(const float* __restrict__ qkv, half_t* __restrict__ out) {
    int lane = threadIdx.x & 31;
    int wid = (blockIdx.x * blockDim.x + threadIdx.x) >> 5;
    int head = wid & 7;
    int pos = (wid >> 3) & (NPOS - 1);
    int bidx = wid >> 15;
    int token = (bidx << 12) | pos;
    int r = pos >> 6, c = pos & 63;

    float q = qkv[(size_t)token * 768 + head * 32 + lane];
    float dots[9], vv[9];
    #pragma unroll
    for (int j = 0; j < 9; ++j) {
        int rr = (r + j / 3 - 1) & 63;
        int cc = (c + j % 3 - 1) & 63;
        int nt = (bidx << 12) | (rr << 6) | cc;
        const float* kv = qkv + (size_t)nt * 768 + head * 32 + lane;
        float d = q * kv[256];
        vv[j] = kv[512];
        dots[j] = wave_sum(d) * 0.17677669529663687f;   // 1/sqrt(32)
    }
    float m = dots[0];
    #pragma unroll
    for (int j = 1; j < 9; ++j) m = fmaxf(m, dots[j]);
    float s = 0.f, e[9];
    #pragma unroll
    for (int j = 0; j < 9; ++j) { e[j] = expf(dots[j] - m); s += e[j]; }
    float inv = 1.0f / s, o = 0.f;
    #pragma unroll
    for (int j = 0; j < 9; ++j) o += e[j] * inv * vv[j];
    out[(size_t)token * DIM + head * 32 + lane] = (half_t)o;
}

// ---------------------------------------------------------------------------
// head: LN + x @ Wh + bh, add to cells[:, PE_IN:], passthrough cells[:, :PE_IN]
// one wave per token.
// ---------------------------------------------------------------------------
__global__ void head_kernel(const float* __restrict__ x,
                            const float* __restrict__ g,
                            const float* __restrict__ bt,
                            const float* __restrict__ Wh,
                            const float* __restrict__ bh,
                            const float* __restrict__ cells,
                            float* __restrict__ out) {
    int lane = threadIdx.x & 31;
    int tok = (blockIdx.x * blockDim.x + threadIdx.x) >> 5;
    int d0 = lane * 8;
    const float4* xp = (const float4*)(x + (size_t)tok * DIM + d0);
    float4 u0 = xp[0], u1 = xp[1];
    float v[8] = {u0.x, u0.y, u0.z, u0.w, u1.x, u1.y, u1.z, u1.w};
    float s = 0.f, q = 0.f;
    #pragma unroll
    for (int i = 0; i < 8; ++i) { s += v[i]; q += v[i] * v[i]; }
    s = wave_sum(s); q = wave_sum(q);
    float mean = s * (1.0f / DIM);
    float var = q * (1.0f / DIM) - mean * mean;
    float rstd = rsqrtf(var + 1e-5f);
    float xn[8];
    #pragma unroll
    for (int i = 0; i < 8; ++i) {
        int d = d0 + i;
        xn[i] = (v[i] - mean) * rstd * g[d] + bt[d];
    }
    float myupd = 0.f;
    #pragma unroll
    for (int ch = 0; ch < UPDATE_DIM; ++ch) {
        float p = 0.f;
        #pragma unroll
        for (int i = 0; i < 8; ++i) p += xn[i] * Wh[(size_t)(d0 + i) * UPDATE_DIM + ch];
        p = wave_sum(p) + bh[ch];
        if (lane == ch + PE_IN) myupd = p;
    }
    int pos = tok & (NPOS - 1);
    int bidx = tok >> 12;
    if (lane < CELL_DIM) {
        size_t idx = ((size_t)(bidx * CELL_DIM + lane)) * NPOS + pos;
        float cv = cells[idx];
        out[idx] = (lane >= PE_IN) ? (cv + myupd) : cv;
    }
}

// ---------------------------------------------------------------------------
// launch
// ---------------------------------------------------------------------------
extern "C" void kernel_launch(void* const* d_in, const int* in_sizes, int n_in,
                              void* d_out, int out_size, void* d_ws, size_t ws_size,
                              hipStream_t stream) {
    const float* cells = (const float*)d_in[0];
    const float* We    = (const float*)d_in[1];
    const float* be    = (const float*)d_in[2];
    const float* ln1_g = (const float*)d_in[3];
    const float* ln1_b = (const float*)d_in[4];
    const float* Wqkv  = (const float*)d_in[5];
    const float* Wo    = (const float*)d_in[6];
    const float* bo    = (const float*)d_in[7];
    const float* ln2_g = (const float*)d_in[8];
    const float* ln2_b = (const float*)d_in[9];
    const float* W1    = (const float*)d_in[10];
    const float* b1    = (const float*)d_in[11];
    const float* W2    = (const float*)d_in[12];
    const float* b2    = (const float*)d_in[13];
    const float* lnh_g = (const float*)d_in[14];
    const float* lnh_b = (const float*)d_in[15];
    const float* Wh    = (const float*)d_in[16];
    const float* bh    = (const float*)d_in[17];
    float* out = (float*)d_out;

    // workspace carve-up (all offsets 256B aligned)
    char* base = (char*)d_ws;
    float*  x     = (float*)(base);                              // 32 MB
    half_t* yh    = (half_t*)(base + 33554432);                  // 16 MB
    half_t* ah    = (half_t*)(base + 33554432 + 16777216);       // 16 MB
    float*  qkv   = (float*)(base + 67108864);                   // 96 MB (qkv f32)
    half_t* hidh  = (half_t*)(base + 67108864);                  // aliases qkv (64 MB)
    char*   wbase = base + 167772160;
    half_t* wqkvT = (half_t*)(wbase);                            // 2 x 768x256
    half_t* woT   = (half_t*)(wbase + 786432);                   // 2 x 256x256
    half_t* w1T   = (half_t*)(wbase + 786432 + 262144);          // 2 x 1024x256
    half_t* w2T   = (half_t*)(wbase + 786432 + 262144 + 1048576);// 2 x 256x1024

    // 1) cast+transpose weights to f16 (Bt layout: [N][K])
    for (int l = 0; l < 2; ++l) {
        castT_kernel<<<(256 * 768 + 255) / 256, 256, 0, stream>>>(
            Wqkv + (size_t)l * 256 * 768, wqkvT + (size_t)l * 768 * 256, 256, 768);
        castT_kernel<<<(256 * 256 + 255) / 256, 256, 0, stream>>>(
            Wo + (size_t)l * 256 * 256, woT + (size_t)l * 256 * 256, 256, 256);
        castT_kernel<<<(256 * 1024 + 255) / 256, 256, 0, stream>>>(
            W1 + (size_t)l * 256 * 1024, w1T + (size_t)l * 1024 * 256, 256, 1024);
        castT_kernel<<<(1024 * 256 + 255) / 256, 256, 0, stream>>>(
            W2 + (size_t)l * 1024 * 256, w2T + (size_t)l * 256 * 1024, 1024, 256);
    }

    // 2) embedding + positional encoding
    embed_kernel<<<(NTOK * 64) / 256, 256, 0, stream>>>(cells, We, be, x);

    // 3) transformer layers
    for (int l = 0; l < 2; ++l) {
        ln_cast_kernel<<<NTOK / 8, 256, 0, stream>>>(x, ln1_g + l * DIM, ln1_b + l * DIM, yh);

        gemm_wmma_kernel<0><<<dim3(768 / 128, NTOK / 64), 256, 0, stream>>>(
            yh, wqkvT + (size_t)l * 768 * 256, nullptr, qkv, nullptr, NTOK, 768, 256);

        attn_kernel<<<(NTOK * HEADS * 32) / 256, 256, 0, stream>>>(qkv, ah);

        gemm_wmma_kernel<1><<<dim3(256 / 128, NTOK / 64), 256, 0, stream>>>(
            ah, woT + (size_t)l * 256 * 256, bo + l * DIM, x, nullptr, NTOK, 256, 256);

        ln_cast_kernel<<<NTOK / 8, 256, 0, stream>>>(x, ln2_g + l * DIM, ln2_b + l * DIM, yh);

        gemm_wmma_kernel<2><<<dim3(1024 / 128, NTOK / 64), 256, 0, stream>>>(
            yh, w1T + (size_t)l * 1024 * 256, b1 + l * MLP_DIM, nullptr, hidh,
            NTOK, 1024, 256);

        gemm_wmma_kernel<1><<<dim3(256 / 128, NTOK / 64), 256, 0, stream>>>(
            hidh, w2T + (size_t)l * 256 * 1024, b2 + l * DIM, x, nullptr, NTOK, 256, 1024);
    }

    // 4) head + cell update
    head_kernel<<<NTOK / 8, 256, 0, stream>>>(x, lnh_g, lnh_b, Wh, bh, cells, out);

    (void)in_sizes; (void)n_in; (void)out_size; (void)ws_size;
}
